// FeatureCorrelator_19207093748303
// MI455X (gfx1250) — matrix operand order repaired and checked by reading, not compiled, not run on hardware
//
#include <hip/hip_runtime.h>

#define KNN 16

typedef __attribute__((ext_vector_type(16))) __bf16 bf16x16;
typedef __attribute__((ext_vector_type(8)))  float  f32x8;

union ABfrag { uint4 u[2]; bf16x16 v; };

__device__ __forceinline__ unsigned short f2bf(float f) {
  union { float f; unsigned u; } x{f};
  unsigned r = x.u + 0x7fffu + ((x.u >> 16) & 1u);
  return (unsigned short)(r >> 16);
}
__device__ __forceinline__ float bf2f(unsigned short h) {
  union { unsigned u; float f; } x{((unsigned)h) << 16};
  return x.f;
}

// ---------------------------------------------------------------------------
// KNN: one thread per query point, brute-force scan with 16-deep insertion.
// ---------------------------------------------------------------------------
__global__ __launch_bounds__(256) void knn_kernel(
    const float* __restrict__ q, const float* __restrict__ r,
    int* __restrict__ idx_out, int B_, int N_)
{
  const int t = blockIdx.x * blockDim.x + threadIdx.x;
  if (t >= B_ * N_) return;
  const int b = t / N_, n = t % N_;
  const float* qb = q + (size_t)b * 3 * N_;
  const float* rb = r + (size_t)b * 3 * N_;
  const float qx = qb[n], qy = qb[N_ + n], qz = qb[2 * N_ + n];
  float best[KNN]; int bi[KNN];
#pragma unroll
  for (int i = 0; i < KNN; ++i) { best[i] = 3.0e38f; bi[i] = 0; }
  for (int j = 0; j < N_; ++j) {
    const float dx = rb[j] - qx, dy = rb[N_ + j] - qy, dz = rb[2 * N_ + j] - qz;
    const float d2 = dx * dx + dy * dy + dz * dz;
    if (d2 < best[KNN - 1]) {
      int p = KNN - 1;
      while (p > 0 && best[p - 1] > d2) { best[p] = best[p - 1]; bi[p] = bi[p - 1]; --p; }
      best[p] = d2; bi[p] = j;
    }
  }
#pragma unroll
  for (int i = 0; i < KNN; ++i) idx_out[(size_t)t * KNN + i] = bi[i];
}

// dir[row] = ref[idx[row]] - query[n]
__global__ __launch_bounds__(256) void make_dir(
    const float* __restrict__ pq, const float* __restrict__ pr,
    const int* __restrict__ idx, float* __restrict__ dir, int B_, int N_)
{
  const int t = blockIdx.x * blockDim.x + threadIdx.x;
  if (t >= B_ * N_ * KNN) return;
  const int b = t / (N_ * KNN);
  const int n = (t / KNN) % N_;
  const int r = idx[t];
  const float* qb = pq + (size_t)b * 3 * N_;
  const float* rb = pr + (size_t)b * 3 * N_;
  dir[(size_t)t * 3 + 0] = rb[r] - qb[n];
  dir[(size_t)t * 3 + 1] = rb[N_ + r] - qb[N_ + n];
  dir[(size_t)t * 3 + 2] = rb[2 * N_ + r] - qb[2 * N_ + n];
}

// [B,C,N] f32 -> [B,N,C] bf16
__global__ __launch_bounds__(256) void transpose_cast(
    const float* __restrict__ in, unsigned short* __restrict__ out,
    int B_, int C_, int N_)
{
  const int t = blockIdx.x * blockDim.x + threadIdx.x;
  if (t >= B_ * C_ * N_) return;
  const int b = t / (C_ * N_);
  const int rem = t % (C_ * N_);
  const int n = rem / C_, c = rem % C_;
  out[t] = f2bf(in[(size_t)b * C_ * N_ + (size_t)c * N_ + n]);
}

// strided f32 weight slice -> contiguous bf16 [rows, cols]
__global__ __launch_bounds__(256) void cast_w(
    const float* __restrict__ src, unsigned short* __restrict__ dst,
    int rows, int cols, int sstride, int coff)
{
  const int t = blockIdx.x * blockDim.x + threadIdx.x;
  if (t >= rows * cols) return;
  const int rr = t / cols, cc = t % cols;
  dst[t] = f2bf(src[(size_t)rr * sstride + coff + cc]);
}

__global__ void zero_f32(float* p, int n) {
  const int t = blockIdx.x * blockDim.x + threadIdx.x;
  if (t < n) p[t] = 0.0f;
}

// ---------------------------------------------------------------------------
// bf16 WMMA GEMM, register-resident weights:
//   C[m,o] = sum_k A[m,k] * W[o,k]     (A,W bf16 row-major, Kd == 256)
// One wave owns a 32-wide N strip. It preloads ALL B fragments for the strip
// across K=256 into registers (2 n-tiles x 8 k-steps x 8 VGPRs = 128 VGPRs),
// then sweeps MT consecutive 16-row M tiles. Per tile: one clause of 16
// global_load_b128 for the A fragments, then 16 WMMAs that only wait on the
// A stream (B never touches memory again). Next tile's A rows are prefetched
// (global_prefetch) while the current tile computes.
// ---------------------------------------------------------------------------
template <int BIAS, int LRELU, int OUTBF16, int MT>
__global__ __launch_bounds__(128) void gemm_wmma_bf16_regB(
    const unsigned short* __restrict__ A,
    const unsigned short* __restrict__ W,
    const float* __restrict__ bias,
    void* __restrict__ Cout,
    int M, int Nout)
{
  constexpr int KD = 256;       // fixed inner dimension
  constexpr int KSTEPS = KD / 32;

  const int lane = threadIdx.x & 31;
  const int wid  = blockIdx.x * (blockDim.x >> 5) + (threadIdx.x >> 5);
  const int nstrips = Nout >> 5;               // 32-wide N strips
  const int mg  = wid / nstrips;               // M-tile group
  const int tn  = (wid % nstrips) << 5;
  const int tm0 = mg * MT;                     // first 16-row M tile
  if (tm0 * 16 >= M) return;
  const int m    = lane & 15;
  const int half = lane >> 4;

  // --- preload B fragments for the whole K range (register resident) ---
  bf16x16 Bf[2][KSTEPS];
#pragma unroll
  for (int t = 0; t < 2; ++t) {
    const unsigned short* brow = W + (size_t)(tn + t * 16 + m) * KD;
#pragma unroll
    for (int ks = 0; ks < KSTEPS; ++ks) {
      ABfrag b;
      b.u[0] = *reinterpret_cast<const uint4*>(brow + ks * 32 + 8 * half);
      b.u[1] = *reinterpret_cast<const uint4*>(brow + ks * 32 + 16 + 8 * half);
      Bf[t][ks] = b.v;
    }
  }

  for (int mt = 0; mt < MT; ++mt) {
    const unsigned short* arow = A + (size_t)((tm0 + mt) * 16 + m) * KD;

    // prefetch next tile's A row into cache while this tile computes
    if (mt + 1 < MT)
      __builtin_prefetch(arow + 16 * KD, 0, 3);

    // one clause of A-fragment loads for all 8 k-steps
    ABfrag a[KSTEPS];
#pragma unroll
    for (int ks = 0; ks < KSTEPS; ++ks) {
      a[ks].u[0] = *reinterpret_cast<const uint4*>(arow + ks * 32 + 8 * half);
      a[ks].u[1] = *reinterpret_cast<const uint4*>(arow + ks * 32 + 16 + 8 * half);
    }

    f32x8 acc0 = {}, acc1 = {};
#pragma unroll
    for (int ks = 0; ks < KSTEPS; ++ks) {
      acc0 = __builtin_amdgcn_wmma_f32_16x16x32_bf16(
          false, a[ks].v, false, Bf[0][ks], (short)0, acc0, false, false);
      acc1 = __builtin_amdgcn_wmma_f32_16x16x32_bf16(
          false, a[ks].v, false, Bf[1][ks], (short)0, acc1, false, false);
    }

    // epilogue
#pragma unroll
    for (int t = 0; t < 2; ++t) {
      const f32x8 acc = t ? acc1 : acc0;
      const int col = tn + t * 16 + m;
      const float bv = BIAS ? bias[col] : 0.0f;
#pragma unroll
      for (int j = 0; j < 8; ++j) {
        float v = acc[j] + bv;
        if (LRELU) v = v > 0.0f ? v : 0.1f * v;
        const size_t off = (size_t)((tm0 + mt) * 16 + 8 * half + j) * Nout + col;
        if (OUTBF16) ((unsigned short*)Cout)[off] = f2bf(v);
        else         ((float*)Cout)[off] = v;
      }
    }
  }
}

// h0[row,c] = lrelu(Y1[bn,c] + Y2[b, idx[row], c] + dir . W0c[c] + b0[c])
__global__ __launch_bounds__(256) void combine_h0(
    const float* __restrict__ Y1, const float* __restrict__ Y2,
    const int* __restrict__ idx, const float* __restrict__ dir,
    const float* __restrict__ w0, const float* __restrict__ b0,
    unsigned short* __restrict__ h0, int N_, int C_, int inch)
{
  const int row = blockIdx.x;       // [0, B*N*K)
  const int c   = threadIdx.x;      // [0, C)
  const int bn  = row / KNN;
  const int b   = bn / N_;
  const int r   = idx[row];
  const float dx = dir[(size_t)row * 3 + 0];
  const float dy = dir[(size_t)row * 3 + 1];
  const float dz = dir[(size_t)row * 3 + 2];
  const float* wc = w0 + (size_t)c * inch + (inch - 3);
  float v = Y1[(size_t)bn * C_ + c] + Y2[((size_t)b * N_ + r) * C_ + c]
          + dx * wc[0] + dy * wc[1] + dz * wc[2] + b0[c];
  v = v > 0.0f ? v : 0.1f * v;
  h0[(size_t)row * C_ + c] = f2bf(v);
}

// Batch stats of the 8-ch WeightNet hidden over all rows (block-reduced).
__global__ __launch_bounds__(256) void wn_stats(
    const float* __restrict__ dir, const float* __restrict__ w1,
    const float* __restrict__ b1, float* __restrict__ sums, int Mrows)
{
  __shared__ float s[16];
  if (threadIdx.x < 16) s[threadIdx.x] = 0.0f;
  __syncthreads();
  const int t = blockIdx.x * blockDim.x + threadIdx.x;
  if (t < Mrows) {
    const float dx = dir[(size_t)t * 3 + 0];
    const float dy = dir[(size_t)t * 3 + 1];
    const float dz = dir[(size_t)t * 3 + 2];
#pragma unroll
    for (int j = 0; j < 8; ++j) {
      const float h = dx * w1[j * 3] + dy * w1[j * 3 + 1] + dz * w1[j * 3 + 2] + b1[j];
      atomicAdd(&s[j], h);
      atomicAdd(&s[8 + j], h * h);
    }
  }
  __syncthreads();
  if (threadIdx.x < 16) atomicAdd(&sums[threadIdx.x], s[threadIdx.x]);
}

// xsum[bn,c] = sum_k wn1(dir1[bn,k]) . x[bn*K+k, c]
__global__ __launch_bounds__(256) void stage1_reduce(
    const unsigned short* __restrict__ x, const float* __restrict__ dir,
    const float* __restrict__ w1, const float* __restrict__ b1,
    const float* __restrict__ gamma, const float* __restrict__ beta,
    const float* __restrict__ w2, const float* __restrict__ b2,
    const float* __restrict__ stats, int Mrows,
    float* __restrict__ xsum, int Cc)
{
  const int bn = blockIdx.x;
  const int c  = threadIdx.x;
  const float invM = 1.0f / (float)Mrows;
  float mj[8], sj[8], bj[8], wx[8], wy[8], wz[8], b1j[8], w2c[8];
#pragma unroll
  for (int j = 0; j < 8; ++j) {
    const float mean = stats[j] * invM;
    const float var  = stats[8 + j] * invM - mean * mean;
    mj[j] = mean;
    sj[j] = rsqrtf(var + 1e-5f) * gamma[j];
    bj[j] = beta[j];
    wx[j] = w1[j * 3]; wy[j] = w1[j * 3 + 1]; wz[j] = w1[j * 3 + 2];
    b1j[j] = b1[j];
    w2c[j] = w2[(size_t)c * 8 + j];
  }
  const float b2c = b2[c];
  float acc = 0.0f;
  for (int k = 0; k < KNN; ++k) {
    const int row = bn * KNN + k;
    const float dx = dir[(size_t)row * 3 + 0];
    const float dy = dir[(size_t)row * 3 + 1];
    const float dz = dir[(size_t)row * 3 + 2];
    float w = b2c;
#pragma unroll
    for (int j = 0; j < 8; ++j) {
      float h = (dx * wx[j] + dy * wy[j] + dz * wz[j] + b1j[j] - mj[j]) * sj[j] + bj[j];
      h = h > 0.0f ? h : 0.0f;
      w += h * w2c[j];
    }
    acc += w * bf2f(x[(size_t)row * Cc + c]);
  }
  xsum[(size_t)bn * Cc + c] = acc;
}

// out[b,c,n] = sum_k wn2(dir2[bn,k]) . xsum[b, idx2[bn,k], c]
__global__ __launch_bounds__(256) void stage2_reduce(
    const float* __restrict__ xsum, const int* __restrict__ idx,
    const float* __restrict__ dir,
    const float* __restrict__ w1, const float* __restrict__ b1,
    const float* __restrict__ gamma, const float* __restrict__ beta,
    const float* __restrict__ w2, const float* __restrict__ b2,
    const float* __restrict__ stats, int Mrows,
    float* __restrict__ out, int Nn, int Cc)
{
  const int bn = blockIdx.x;
  const int b  = bn / Nn;
  const int n  = bn % Nn;
  const int c  = threadIdx.x;
  const float invM = 1.0f / (float)Mrows;
  float mj[8], sj[8], bj[8], wx[8], wy[8], wz[8], b1j[8], w2c[8];
#pragma unroll
  for (int j = 0; j < 8; ++j) {
    const float mean = stats[j] * invM;
    const float var  = stats[8 + j] * invM - mean * mean;
    mj[j] = mean;
    sj[j] = rsqrtf(var + 1e-5f) * gamma[j];
    bj[j] = beta[j];
    wx[j] = w1[j * 3]; wy[j] = w1[j * 3 + 1]; wz[j] = w1[j * 3 + 2];
    b1j[j] = b1[j];
    w2c[j] = w2[(size_t)c * 8 + j];
  }
  const float b2c = b2[c];
  float acc = 0.0f;
  for (int k = 0; k < KNN; ++k) {
    const int row = bn * KNN + k;
    const int r   = idx[row];
    const float dx = dir[(size_t)row * 3 + 0];
    const float dy = dir[(size_t)row * 3 + 1];
    const float dz = dir[(size_t)row * 3 + 2];
    float w = b2c;
#pragma unroll
    for (int j = 0; j < 8; ++j) {
      float h = (dx * wx[j] + dy * wy[j] + dz * wz[j] + b1j[j] - mj[j]) * sj[j] + bj[j];
      h = h > 0.0f ? h : 0.0f;
      w += h * w2c[j];
    }
    acc += w * xsum[((size_t)b * Nn + r) * Cc + c];
  }
  out[(size_t)b * Cc * Nn + (size_t)c * Nn + n] = acc;
}

// ---------------------------------------------------------------------------
extern "C" void kernel_launch(void* const* d_in, const int* in_sizes, int n_in,
                              void* d_out, int out_size, void* d_ws, size_t ws_size,
                              hipStream_t stream) {
  constexpr int B = 2, N = 4096, C = 256, INCH = 515;
  constexpr int BN = B * N;         // 8192
  constexpr int M  = BN * KNN;      // 131072
  constexpr int MT = 4;             // M tiles per wave in the GEMM
  (void)in_sizes; (void)n_in; (void)out_size; (void)ws_size;

  const float* pc1      = (const float*)d_in[0];
  const float* pc2      = (const float*)d_in[1];
  const float* feat1    = (const float*)d_in[2];
  const float* feat2    = (const float*)d_in[3];
  const float* mlp_w0   = (const float*)d_in[4];
  const float* mlp_b0   = (const float*)d_in[5];
  const float* mlp_w1   = (const float*)d_in[6];
  const float* mlp_b1   = (const float*)d_in[7];
  const float* mlp_w2   = (const float*)d_in[8];
  const float* mlp_b2   = (const float*)d_in[9];
  const float* wn1_w1   = (const float*)d_in[10];
  const float* wn1_b1   = (const float*)d_in[11];
  const float* wn1_gam  = (const float*)d_in[12];
  const float* wn1_bet  = (const float*)d_in[13];
  const float* wn1_w2   = (const float*)d_in[14];
  const float* wn1_b2   = (const float*)d_in[15];
  const float* wn2_w1   = (const float*)d_in[16];
  const float* wn2_b1   = (const float*)d_in[17];
  const float* wn2_gam  = (const float*)d_in[18];
  const float* wn2_bet  = (const float*)d_in[19];
  const float* wn2_w2   = (const float*)d_in[20];
  const float* wn2_b2   = (const float*)d_in[21];

  char* p = (char*)d_ws;
  auto alloc = [&](size_t bytes) -> void* {
    void* r = (void*)p;
    p += (bytes + 255) & ~(size_t)255;
    return r;
  };
  int*            idx1 = (int*)alloc(sizeof(int) * (size_t)M);
  int*            idx2 = (int*)alloc(sizeof(int) * (size_t)M);
  float*          dir1 = (float*)alloc(sizeof(float) * (size_t)M * 3);
  float*          dir2 = (float*)alloc(sizeof(float) * (size_t)M * 3);
  unsigned short* F1t  = (unsigned short*)alloc(2ull * BN * C);
  unsigned short* F2t  = (unsigned short*)alloc(2ull * BN * C);
  unsigned short* Wa   = (unsigned short*)alloc(2ull * C * C);
  unsigned short* Wb   = (unsigned short*)alloc(2ull * C * C);
  unsigned short* W1b  = (unsigned short*)alloc(2ull * C * C);
  unsigned short* W2b  = (unsigned short*)alloc(2ull * C * C);
  float*          Y1   = (float*)alloc(4ull * BN * C);
  float*          Y2   = (float*)alloc(4ull * BN * C);
  unsigned short* h0   = (unsigned short*)alloc(2ull * M * C);  // also holds h2
  unsigned short* h1   = (unsigned short*)alloc(2ull * M * C);
  float*          xsum = (float*)alloc(4ull * BN * C);
  float*          stats= (float*)alloc(4ull * 32);

  // prep
  zero_f32<<<1, 32, 0, stream>>>(stats, 32);
  transpose_cast<<<(B * C * N + 255) / 256, 256, 0, stream>>>(feat1, F1t, B, C, N);
  transpose_cast<<<(B * C * N + 255) / 256, 256, 0, stream>>>(feat2, F2t, B, C, N);
  cast_w<<<(C * C + 255) / 256, 256, 0, stream>>>(mlp_w0, Wa, C, C, INCH, 0);
  cast_w<<<(C * C + 255) / 256, 256, 0, stream>>>(mlp_w0, Wb, C, C, INCH, C);
  cast_w<<<(C * C + 255) / 256, 256, 0, stream>>>(mlp_w1, W1b, C, C, C, 0);
  cast_w<<<(C * C + 255) / 256, 256, 0, stream>>>(mlp_w2, W2b, C, C, C, 0);

  // KNN + directions
  knn_kernel<<<(BN + 255) / 256, 256, 0, stream>>>(pc1, pc2, idx1, B, N);
  knn_kernel<<<(BN + 255) / 256, 256, 0, stream>>>(pc1, pc1, idx2, B, N);
  make_dir<<<(M + 255) / 256, 256, 0, stream>>>(pc1, pc2, idx1, dir1, B, N);
  make_dir<<<(M + 255) / 256, 256, 0, stream>>>(pc1, pc1, idx2, dir2, B, N);

  // layer0 factored: Y1 = F1.W0a^T, Y2 = F2.W0b^T (gather commutes with GEMM)
  {
    const int waves  = (BN / 16 / MT) * (C / 32);
    const int blocks = waves / 4;
    gemm_wmma_bf16_regB<0, 0, 0, MT><<<blocks, 128, 0, stream>>>(F1t, Wa, nullptr, Y1, BN, C);
    gemm_wmma_bf16_regB<0, 0, 0, MT><<<blocks, 128, 0, stream>>>(F2t, Wb, nullptr, Y2, BN, C);
  }
  combine_h0<<<M, C, 0, stream>>>(Y1, Y2, idx1, dir1, mlp_w0, mlp_b0, h0, N, C, INCH);

  // layers 1 & 2 — the heavy WMMA GEMMs (131072 x 256 x 256 each)
  {
    const int waves  = (M / 16 / MT) * (C / 32);
    const int blocks = waves / 4;
    gemm_wmma_bf16_regB<1, 1, 1, MT><<<blocks, 128, 0, stream>>>(h0, W1b, mlp_b1, h1, M, C);
    gemm_wmma_bf16_regB<1, 1, 1, MT><<<blocks, 128, 0, stream>>>(h1, W2b, mlp_b2, h0, M, C);
  }

  // WeightNet batch stats
  wn_stats<<<(M + 255) / 256, 256, 0, stream>>>(dir1, wn1_w1, wn1_b1, stats, M);
  wn_stats<<<(M + 255) / 256, 256, 0, stream>>>(dir2, wn2_w1, wn2_b1, stats + 16, M);

  // weighted K-reductions
  stage1_reduce<<<BN, C, 0, stream>>>(h0, dir1, wn1_w1, wn1_b1, wn1_gam, wn1_bet,
                                      wn1_w2, wn1_b2, stats, M, xsum, C);
  stage2_reduce<<<BN, C, 0, stream>>>(xsum, idx2, dir2, wn2_w1, wn2_b1, wn2_gam,
                                      wn2_bet, wn2_w2, wn2_b2, stats + 16, M,
                                      (float*)d_out, N, C);
}
